// TransformerPredictor_22247930593971
// MI455X (gfx1250) — matrix operand order repaired and checked by reading, not compiled, not run on hardware
//
#include <hip/hip_runtime.h>
#include <hip/hip_bf16.h>

typedef _Float16 half16 __attribute__((ext_vector_type(16)));
typedef _Float16 half8  __attribute__((ext_vector_type(8)));
typedef float    float8 __attribute__((ext_vector_type(8)));

#define HIDN   512
#define POOLN  4096
#define NOBJ   1024
#define NRELN  128
#define NOBJC  151
#define NRELC  51
#define EMBD   200

__device__ __forceinline__ void cvt4(half16& f, int base, float4 v) {
    f[base + 0] = (_Float16)v.x; f[base + 1] = (_Float16)v.y;
    f[base + 2] = (_Float16)v.z; f[base + 3] = (_Float16)v.w;
}

// ---------------------------------------------------------------------------
// WMMA GEMM: C[M,N] = A[M,K] x B + bias, optional ReLU.  (f32 in/out)
// Requirements (guaranteed by call sites): K%32==0, M%16==0, exact grid.
// N may be ragged (loads clamped, stores guarded).
// TRANSB: B is (N,K) row-major; else (K,N). Non-transposed B staged through
// double-buffered LDS shared by the 8 waves of the block.
// ---------------------------------------------------------------------------
template <bool TRANSB, bool RELU>
__global__ __launch_bounds__(256) void wmma_gemm(
    const float* __restrict__ A, const float* __restrict__ B,
    const float* __restrict__ bias, float* __restrict__ C,
    int M, int N, int K, int lda, int ldb, int ldc)
{
    __shared__ __align__(16) _Float16 bsh[2][16 * 32];

    const int lane = threadIdx.x & 31;
    const int wave = threadIdx.x >> 5;
    const int m0 = (blockIdx.y * 8 + wave) * 16;
    const int n0 = blockIdx.x * 16;
    const int ml  = lane & 15;
    const int khf = lane >> 4;

    const int arow  = m0 + ml;
    const int bcol  = n0 + ml;
    const int bcolc = bcol < N ? bcol : N - 1;

    float8 acc = {};
    int ib = 0;
    for (int kk = 0; kk < K; kk += 32, ib ^= 1) {
        const float* ab = A + (long)arow * lda + kk + (khf << 4);
        __builtin_prefetch((const void*)(ab + 32), 0, 0);   // next K-chunk
        const float4 a0 = *(const float4*)(ab);
        const float4 a1 = *(const float4*)(ab + 4);
        const float4 a2 = *(const float4*)(ab + 8);
        const float4 a3 = *(const float4*)(ab + 12);
        half16 af;
        cvt4(af, 0, a0); cvt4(af, 4, a1); cvt4(af, 8, a2); cvt4(af, 12, a3);

        half16 bf;
        if (TRANSB) {
            const float* bb = B + (long)bcolc * ldb + kk + (khf << 4);
            const float4 b0 = *(const float4*)(bb);
            const float4 b1 = *(const float4*)(bb + 4);
            const float4 b2 = *(const float4*)(bb + 8);
            const float4 b3 = *(const float4*)(bb + 12);
            cvt4(bf, 0, b0); cvt4(bf, 4, b1); cvt4(bf, 8, b2); cvt4(bf, 12, b3);
        } else {
            _Float16* buf = bsh[ib];
#pragma unroll
            for (int e = 0; e < 2; ++e) {
                const int idx  = (int)threadIdx.x + e * 256;
                const int col  = idx & 15;
                const int krow = idx >> 4;
                int gcol = n0 + col; gcol = gcol < N ? gcol : N - 1;
                buf[col * 32 + krow] = (_Float16)B[(long)(kk + krow) * ldb + gcol];
            }
            __syncthreads();
            const _Float16* src = buf + ml * 32 + (khf << 4);
            const half8 b0 = *(const half8*)(src);
            const half8 b1 = *(const half8*)(src + 8);
#pragma unroll
            for (int j = 0; j < 8; ++j) { bf[j] = b0[j]; bf[8 + j] = b1[j]; }
        }

        acc = __builtin_amdgcn_wmma_f32_16x16x32_f16(
            false, af, false, bf, (short)0, acc, false, false);
    }

    if (bcol < N) {
        const float bv = bias[bcol];
#pragma unroll
        for (int v = 0; v < 8; ++v) {
            float val = acc[v] + bv;
            if (RELU) val = fmaxf(val, 0.f);
            C[(long)(m0 + v + (khf << 3)) * ldc + bcol] = val;
        }
    }
}

// ---------------------------------------------------------------------------
// conv1: Cin=1, 5x5 pad2, relu + fused 2x2 mean pool.
// Input (NB,200,200) f32; weights pre-packed f16 (64,32) (K padded 25->32
// with zeros, so A needs no K guard); output pooled NHWC f16 (NB,100,100,64).
// One A gather per wave feeds ALL 4 Cout tiles (Cout=64): 4 wmmas per gather,
// activation traffic /4.  grid.x == 1.
// ---------------------------------------------------------------------------
__global__ __launch_bounds__(256) void wmma_conv1(
    const float* __restrict__ in, const _Float16* __restrict__ wgt,
    const float* __restrict__ bias, _Float16* __restrict__ out,
    int NB, int H, int W, int Cout)
{
    const int lane = threadIdx.x & 31;
    const int wave = threadIdx.x >> 5;
    const int Mtot = NB * H * W;
    const int m0 = (blockIdx.y * 8 + wave) * 16;
    if (m0 >= Mtot) return;

    const int ml  = lane & 15;
    const int khf = lane >> 4;
    const int PH = H >> 1, PW = W >> 1;

    const int arow = m0 + ml;
    const int q = arow >> 2, s4 = arow & 3;
    const int n = q / (PH * PW);
    const int rq = q - n * (PH * PW);
    const int ph = rq / PW;
    const int oh = (ph << 1) + (s4 >> 1);
    const int ow = ((rq - ph * PW) << 1) + (s4 & 1);
    const float* inb = in + (long)n * H * W;

    half16 af;
#pragma unroll
    for (int v = 0; v < 8; ++v) {
#pragma unroll
        for (int e = 0; e < 2; ++e) {
            int k = (khf << 4) + (v << 1) + e;
            k = k < 25 ? k : 24;            // padded weights are 0 beyond 24
            const int ih = oh + k / 5 - 2;
            const int iw = ow + (k % 5) - 2;
            const bool valid = (ih >= 0) & (ih < H) & (iw >= 0) & (iw < W);
            const int ihc = ih < 0 ? 0 : (ih >= H ? H - 1 : ih);
            const int iwc = iw < 0 ? 0 : (iw >= W ? W - 1 : iw);
            const float x = inb[ihc * W + iwc];
            af[2 * v + e] = (_Float16)(valid ? x : 0.f);
        }
    }

    float8 acc[4];
#pragma unroll
    for (int t = 0; t < 4; ++t) {
        const int co = ml + 16 * t;
        const _Float16* wrow = wgt + co * 32 + (khf << 4);
        const half8 b0 = *(const half8*)(wrow);
        const half8 b1 = *(const half8*)(wrow + 8);
        half16 bf;
#pragma unroll
        for (int j = 0; j < 8; ++j) { bf[j] = b0[j]; bf[8 + j] = b1[j]; }
        float8 z = {};
        acc[t] = __builtin_amdgcn_wmma_f32_16x16x32_f16(
            false, af, false, bf, (short)0, z, false, false);
    }

#pragma unroll
    for (int t = 0; t < 4; ++t) {
        const int co = ml + 16 * t;
        const float bv = bias[co];
#pragma unroll
        for (int g = 0; g < 2; ++g) {
            float s = 0.f;
#pragma unroll
            for (int e = 0; e < 4; ++e) s += fmaxf(acc[t][4 * g + e] + bv, 0.f);
            s *= 0.25f;
            const long qd = (m0 >> 2) + khf * 2 + g;     // pooled NHWC pixel
            out[qd * Cout + co] = (_Float16)s;
        }
    }
}

// ---------------------------------------------------------------------------
// NHWC fp16 implicit-GEMM conv, CIN % 32 == 0 (template constant).
// Each 32-wide K-chunk = one (kh,kw) tap x 32 contiguous channels:
// A fragment = two b128 half loads + one validity select per tap.
// Weights pre-packed f16 (Cout, KHW*CIN) with ci innermost.
// Each wave computes TWO 16-wide Cout tiles from one A fragment (Cout%32==0):
// 2 wmmas per A load, activation traffic halved.
// POOL: fused relu + 2x2 mean pool; output NHWC f16 (coalesced q*Cout+co).
// ---------------------------------------------------------------------------
template <int KHH, int KWW, int PAD, bool POOL, int CIN>
__global__ __launch_bounds__(256) void wmma_conv_nhwc(
    const _Float16* __restrict__ in, const _Float16* __restrict__ wgt,
    const float* __restrict__ bias, _Float16* __restrict__ out,
    int NB, int H, int W, int Cout)
{
    constexpr int KHW = KHH * KWW;
    const int lane = threadIdx.x & 31;
    const int wave = threadIdx.x >> 5;
    const int Mtot = NB * H * W;
    const int m0 = (blockIdx.y * 8 + wave) * 16;
    const int n0 = blockIdx.x * 32;                  // two 16-wide Cout tiles
    if (m0 >= Mtot) return;

    const int ml  = lane & 15;
    const int khf = lane >> 4;
    const int PH = H >> 1, PW = W >> 1;

    const int arow = m0 + ml;
    int n, oh, ow;
    if (POOL) {
        const int q = arow >> 2, s4 = arow & 3;
        n = q / (PH * PW);
        const int rq = q - n * (PH * PW);
        const int ph = rq / PW;
        oh = (ph << 1) + (s4 >> 1);
        ow = ((rq - ph * PW) << 1) + (s4 & 1);
    } else {
        n = arow / (H * W);
        const int r2 = arow - n * (H * W);
        oh = r2 / W;
        ow = r2 - oh * W;
    }
    const _Float16* inb = in + (long)n * H * W * CIN;
    const int co0 = n0 + ml;
    const _Float16* wrow0 = wgt + (long)co0 * (KHW * CIN);
    const _Float16* wrow1 = wrow0 + (long)16 * (KHW * CIN);

    float8 acc0 = {}, acc1 = {};
    for (int plane = 0; plane < KHW; ++plane) {
        const int kh = plane / KWW, kw = plane - (plane / KWW) * KWW;
        const int ih = oh + kh - PAD, iw = ow + kw - PAD;
        const bool valid = (ih >= 0) & (ih < H) & (iw >= 0) & (iw < W);
        const int ihc = valid ? ih : 0, iwc = valid ? iw : 0;
        const _Float16* pb = inb + ((long)ihc * W + iwc) * CIN;
        const _Float16* wb0 = wrow0 + plane * CIN;
        const _Float16* wb1 = wrow1 + plane * CIN;
#pragma unroll
        for (int cc = 0; cc < CIN; cc += 32) {
            const half8 a0 = *(const half8*)(pb + cc + (khf << 4));
            const half8 a1 = *(const half8*)(pb + cc + (khf << 4) + 8);
            half16 af;
#pragma unroll
            for (int j = 0; j < 8; ++j) {
                af[j]     = valid ? a0[j] : (_Float16)0.f;
                af[8 + j] = valid ? a1[j] : (_Float16)0.f;
            }
            const half8 w0 = *(const half8*)(wb0 + cc + (khf << 4));
            const half8 w1 = *(const half8*)(wb0 + cc + (khf << 4) + 8);
            const half8 w2 = *(const half8*)(wb1 + cc + (khf << 4));
            const half8 w3 = *(const half8*)(wb1 + cc + (khf << 4) + 8);
            half16 bf0, bf1;
#pragma unroll
            for (int j = 0; j < 8; ++j) {
                bf0[j] = w0[j]; bf0[8 + j] = w1[j];
                bf1[j] = w2[j]; bf1[8 + j] = w3[j];
            }
            acc0 = __builtin_amdgcn_wmma_f32_16x16x32_f16(
                false, af, false, bf0, (short)0, acc0, false, false);
            acc1 = __builtin_amdgcn_wmma_f32_16x16x32_f16(
                false, af, false, bf1, (short)0, acc1, false, false);
        }
    }

    const float bv0 = bias[co0];
    const float bv1 = bias[co0 + 16];
    if (POOL) {
#pragma unroll
        for (int g = 0; g < 2; ++g) {
            float s0 = 0.f, s1 = 0.f;
#pragma unroll
            for (int e = 0; e < 4; ++e) {
                s0 += fmaxf(acc0[4 * g + e] + bv0, 0.f);
                s1 += fmaxf(acc1[4 * g + e] + bv1, 0.f);
            }
            const long qd = (m0 >> 2) + khf * 2 + g;
            out[qd * Cout + co0]      = (_Float16)(s0 * 0.25f);
            out[qd * Cout + co0 + 16] = (_Float16)(s1 * 0.25f);
        }
    } else {
#pragma unroll
        for (int v = 0; v < 8; ++v) {
            const long r = m0 + v + (khf << 3);
            out[r * Cout + co0]      = (_Float16)fmaxf(acc0[v] + bv0, 0.f);
            out[r * Cout + co0 + 16] = (_Float16)fmaxf(acc1[v] + bv1, 0.f);
        }
    }
}

// ------------------------- weight repack kernels ---------------------------
__global__ void k_pack_w1(const float* __restrict__ w, _Float16* __restrict__ w2)
{   // (64,25) f32 -> (64,32) f16 zero-padded
    const int t = blockIdx.x * blockDim.x + threadIdx.x;
    if (t >= 64 * 32) return;
    const int co = t >> 5, k = t & 31;
    w2[t] = (_Float16)(k < 25 ? w[co * 25 + k] : 0.f);
}

__global__ void k_repack_w(const float* __restrict__ w, _Float16* __restrict__ w2,
                           int Cout, int Cin, int KHW_)
{   // OIHW (Cout,Cin,KHW) f32 -> (Cout, KHW*Cin) f16, ci innermost
    const int t = blockIdx.x * blockDim.x + threadIdx.x;
    const int tot = Cout * Cin * KHW_;
    if (t >= tot) return;
    const int co = t / (Cin * KHW_);
    const int rem = t - co * (Cin * KHW_);
    const int ci = rem / KHW_;
    const int p  = rem - ci * KHW_;
    w2[(long)co * (KHW_ * Cin) + p * Cin + ci] = (_Float16)w[t];
}

// ------------------------- small elementwise kernels ------------------------
__global__ void k_outer(const float* __restrict__ emb, const int* __restrict__ pair,
                        float* __restrict__ x0)
{
    const int t = blockIdx.x * blockDim.x + threadIdx.x;
    if (t >= NRELN * EMBD * EMBD) return;
    const int r = t / (EMBD * EMBD), rem = t % (EMBD * EMBD);
    const int i = rem / EMBD, j = rem % EMBD;
    const int a = pair[2 * r] + 1, b = pair[2 * r + 1] + 1;
    x0[t] = emb[a * EMBD + i] * emb[b * EMBD + j];
}

__global__ void k_gather_prod(const float* __restrict__ edge, const int* __restrict__ pair,
                              float* __restrict__ prod)
{
    const int t = blockIdx.x * blockDim.x + threadIdx.x;
    if (t >= NRELN * 1024) return;
    const int r = t >> 10, c = t & 1023;
    const int idx = (c < HIDN) ? pair[2 * r] : pair[2 * r + 1];
    prod[t] = edge[(long)idx * 1024 + c];
}

__global__ void k_mul(float* __restrict__ a, const float* __restrict__ b, int n)
{
    const int t = blockIdx.x * blockDim.x + threadIdx.x;
    if (t < n) a[t] *= b[t];
}

__global__ void k_relcat(const float* __restrict__ prod, const float* __restrict__ edgef,
                         float* __restrict__ cat)
{
    const int t = blockIdx.x * blockDim.x + threadIdx.x;
    if (t >= NRELN * 1536) return;
    const int r = t / 1536, c = t % 1536;
    float v;
    if (c < 512)        v = prod[r * 1024 + c];
    else if (c < 1024)  v = edgef[r * 512 + (c - 512)];
    else                v = prod[r * 1024 + (c - 512)];
    cat[t] = v;
}

__global__ void k_freqadd(float* __restrict__ outd, const float* __restrict__ freq,
                          const int* __restrict__ preds, const int* __restrict__ pair)
{
    const int t = blockIdx.x * blockDim.x + threadIdx.x;
    if (t >= NRELN * NRELC) return;
    const int r = t / NRELC, c = t % NRELC;
    const int c0 = preds[pair[2 * r]];
    const int c1 = preds[pair[2 * r + 1]];
    outd[t] += freq[((long)c0 * NOBJC + c1) * NRELC + c];
}

__global__ void k_cat2(const float* __restrict__ pwe, const float* __restrict__ uf,
                       float* __restrict__ cat)
{
    const int t = blockIdx.x * blockDim.x + threadIdx.x;
    if (t >= NRELN * 8192) return;
    const int r = t >> 13, c = t & 8191;
    cat[t] = (c < 4096) ? pwe[r * 4096 + c] : uf[r * 4096 + (c - 4096)];
}

__global__ void k_mean_nhwc(const _Float16* __restrict__ x, float* __restrict__ y)
{   // (128, 625, 512) NHWC f16 -> (128,512) f32 mean over spatial
    const int t = blockIdx.x * blockDim.x + threadIdx.x;
    if (t >= NRELN * 512) return;
    const int n = t >> 9, c = t & 511;
    const _Float16* p = x + (long)n * 625 * 512 + c;
    float s = 0.f;
    for (int i = 0; i < 625; ++i) s += (float)p[(long)i * 512];
    y[t] = s * (1.f / 625.f);
}

// ------------------------------- launchers ---------------------------------
static inline void gemm(const float* A, const float* B, const float* bias, float* C,
                        int M, int N, int K, int lda, int ldb, int ldc,
                        bool transb, bool relu, hipStream_t s)
{
    dim3 g((N + 15) / 16, (M + 127) / 128), blk(256);
    if (transb) {
        if (relu) wmma_gemm<true, true><<<g, blk, 0, s>>>(A, B, bias, C, M, N, K, lda, ldb, ldc);
        else      wmma_gemm<true, false><<<g, blk, 0, s>>>(A, B, bias, C, M, N, K, lda, ldb, ldc);
    } else {
        if (relu) wmma_gemm<false, true><<<g, blk, 0, s>>>(A, B, bias, C, M, N, K, lda, ldb, ldc);
        else      wmma_gemm<false, false><<<g, blk, 0, s>>>(A, B, bias, C, M, N, K, lda, ldb, ldc);
    }
}

static inline dim3 conv_grid(int NB, int H, int W, int CoutTiles)
{
    const long Mtot = (long)NB * H * W;
    return dim3(CoutTiles, (unsigned)((Mtot / 16 + 7) / 8));
}

extern "C" void kernel_launch(void* const* d_in, const int* in_sizes, int n_in,
                              void* d_out, int out_size, void* d_ws, size_t ws_size,
                              hipStream_t stream)
{
    (void)in_sizes; (void)n_in; (void)out_size; (void)ws_size;

    const float* obj_feats  = (const float*)d_in[0];
    const int*   obj_preds  = (const int*)  d_in[1];
    const int*   pair_idx   = (const int*)  d_in[2];
    const float* unionf     = (const float*)d_in[3];
    const float* W_post_emb = (const float*)d_in[4];
    const float* b_post_emb = (const float*)d_in[5];
    const float* W_post_cat = (const float*)d_in[6];
    const float* b_post_cat = (const float*)d_in[7];
    const float* W_edge     = (const float*)d_in[8];
    const float* b_edge     = (const float*)d_in[9];
    const float* W_lin_rel  = (const float*)d_in[10];
    const float* b_lin_rel  = (const float*)d_in[11];
    const float* W_rel      = (const float*)d_in[12];
    const float* b_rel      = (const float*)d_in[13];
    const float* freq       = (const float*)d_in[14];
    const float* embed      = (const float*)d_in[15];
    const float* cw1 = (const float*)d_in[16]; const float* cb1 = (const float*)d_in[17];
    const float* cw2 = (const float*)d_in[18]; const float* cb2 = (const float*)d_in[19];
    const float* cw3 = (const float*)d_in[20]; const float* cb3 = (const float*)d_in[21];
    const float* cw4 = (const float*)d_in[22]; const float* cb4 = (const float*)d_in[23];
    const float* cw5 = (const float*)d_in[24]; const float* cb5 = (const float*)d_in[25];
    const float* cw6 = (const float*)d_in[26]; const float* cb6 = (const float*)d_in[27];
    const float* Ww1 = (const float*)d_in[28]; const float* bw1 = (const float*)d_in[29];
    const float* Ww2 = (const float*)d_in[30]; const float* bw2 = (const float*)d_in[31];

    // ---- workspace layout (byte regions, all 64B aligned) ----
    char* base = (char*)d_ws;
    // [0,16MB): f32 scratch
    float* edge_rep = (float*)base;                // 1,048,576 f
    float* prod     = edge_rep + 1048576;          //   131,072 f
    float* ctx      = prod + 131072;               //   524,288 f
    float* edgef    = ctx + 524288;                //    65,536 f
    float* relcat   = edgef + 65536;               //   196,608 f
    float* relf     = relcat + 196608;             //    65,536 f
    float* meanv    = relf + 65536;                //    65,536 f
    float* c5       = meanv + 65536;               //    49,152 f
    float* pwe      = c5 + 49152;                  //   524,288 f
    float* cat2     = pwe + 524288;                // 1,048,576 f
    float* hbuf     = cat2 + 1048576;              //    65,536 f
    // [16MB,20MB): f16 repacked weights
    _Float16* w1p = (_Float16*)(base + (16l << 20));   //     2,048 h
    _Float16* w2p = w1p + 2048;                        //    73,728 h
    _Float16* w3p = w2p + 73728;                       //   294,912 h
    _Float16* w4p = w3p + 294912;                      // 1,179,648 h
    // [20MB,102MB): BIG_A = x0(f32 20.5MB) / c2p(f16 82MB) / c4(f16 82MB)
    char* bigA = base + (20l << 20);
    // [102MB,266MB): BIG_B = c1p(f16 164MB) / c3p(f16 41MB)
    char* bigB = base + (102l << 20);

    float* rel_dists = (float*)d_out;                      // 128*51
    float* union_out = (float*)d_out + NRELN * NRELC;      // 128*4096

    // ================= relational branch =================
    gemm(obj_feats, W_post_emb, b_post_emb, edge_rep, NOBJ, 1024, HIDN, HIDN, 1024, 1024,
         false, false, stream);
    k_gather_prod<<<(NRELN * 1024 + 255) / 256, 256, 0, stream>>>(edge_rep, pair_idx, prod);
    gemm(prod, W_post_cat, b_post_cat, ctx, NRELN, POOLN, 1024, 1024, POOLN, POOLN,
         false, false, stream);
    k_mul<<<(NRELN * POOLN + 255) / 256, 256, 0, stream>>>(ctx, unionf, NRELN * POOLN);
    gemm(ctx, W_edge, b_edge, edgef, NRELN, HIDN, POOLN, POOLN, HIDN, HIDN,
         false, true, stream);
    k_relcat<<<(NRELN * 1536 + 255) / 256, 256, 0, stream>>>(prod, edgef, relcat);
    gemm(relcat, W_lin_rel, b_lin_rel, relf, NRELN, HIDN, 1536, 1536, HIDN, HIDN,
         false, false, stream);
    gemm(relf, W_rel, b_rel, rel_dists, NRELN, NRELC, HIDN, HIDN, NRELC, NRELC,
         false, false, stream);
    k_freqadd<<<(NRELN * NRELC + 255) / 256, 256, 0, stream>>>(rel_dists, freq, obj_preds, pair_idx);

    // ================= CNN branch (NHWC f16 chain) =================
    k_pack_w1<<<(64 * 32 + 255) / 256, 256, 0, stream>>>(cw1, w1p);
    k_repack_w<<<(128 * 64 * 9 + 255) / 256, 256, 0, stream>>>(cw2, w2p, 128, 64, 9);
    k_repack_w<<<(256 * 128 * 9 + 255) / 256, 256, 0, stream>>>(cw3, w3p, 256, 128, 9);
    k_repack_w<<<(512 * 256 * 9 + 255) / 256, 256, 0, stream>>>(cw4, w4p, 512, 256, 9);

    float*     x0  = (float*)bigA;
    _Float16*  c1p = (_Float16*)bigB;   // (128,100,100,64) NHWC
    _Float16*  c2p = (_Float16*)bigA;   // (128,50,50,128)
    _Float16*  c3p = (_Float16*)bigB;   // (128,25,25,256)
    _Float16*  c4  = (_Float16*)bigA;   // (128,25,25,512)

    k_outer<<<(NRELN * EMBD * EMBD + 255) / 256, 256, 0, stream>>>(embed, pair_idx, x0);
    // conv1: all 4 Cout tiles per wave -> grid.x = 1
    wmma_conv1<<<conv_grid(NRELN, 200, 200, 1), 256, 0, stream>>>(
        x0, w1p, cb1, c1p, NRELN, 200, 200, 64);
    // conv2..4: 2 Cout tiles per wave -> grid.x = Cout/32
    wmma_conv_nhwc<3, 3, 1, true, 64><<<conv_grid(NRELN, 100, 100, 128 / 32), 256, 0, stream>>>(
        c1p, w2p, cb2, c2p, NRELN, 100, 100, 128);
    wmma_conv_nhwc<3, 3, 1, true, 128><<<conv_grid(NRELN, 50, 50, 256 / 32), 256, 0, stream>>>(
        c2p, w3p, cb3, c3p, NRELN, 50, 50, 256);
    wmma_conv_nhwc<3, 3, 1, false, 256><<<conv_grid(NRELN, 25, 25, 512 / 32), 256, 0, stream>>>(
        c3p, w4p, cb4, c4, NRELN, 25, 25, 512);
    k_mean_nhwc<<<(NRELN * 512 + 255) / 256, 256, 0, stream>>>(c4, meanv);
    gemm(meanv, cw5, cb5, c5, NRELN, 384, 512, 512, 512, 384, true, true, stream);
    gemm(c5, cw6, cb6, pwe, NRELN, 4096, 384, 384, 384, 4096, true, false, stream);

    // ================= fusion MLP =================
    k_cat2<<<(NRELN * 8192 + 255) / 256, 256, 0, stream>>>(pwe, unionf, cat2);
    gemm(cat2, Ww1, bw1, hbuf, NRELN, HIDN, 8192, 8192, HIDN, HIDN, false, true, stream);
    gemm(hbuf, Ww2, bw2, union_out, NRELN, POOLN, HIDN, HIDN, POOLN, POOLN, false, true, stream);
}
